// SPTnet_19988777796209
// MI455X (gfx1250) — matrix operand order repaired and checked by reading, not compiled, not run on hardware
//
#include <hip/hip_runtime.h>
#include <cstddef>

// ---------------------------------------------------------------------------
// Types for CDNA5 WMMA
// ---------------------------------------------------------------------------
typedef __attribute__((ext_vector_type(16))) __bf16        v16bf;
typedef __attribute__((ext_vector_type(8)))  float         v8f;
typedef __attribute__((ext_vector_type(4)))  unsigned int  u32x4;

typedef unsigned short ushort_t;

__device__ __forceinline__ ushort_t f2bf(float f) {
    unsigned int u = __builtin_bit_cast(unsigned int, f);
    unsigned int r = (u + 0x7FFFu + ((u >> 16) & 1u)) >> 16;   // round-to-nearest-even
    return (ushort_t)r;
}

// ===========================================================================
// Localization conv1: 7x7 VALID (Cin=3 -> 32) + maxpool2 + relu
// in : ori_img [16][3][224][224]   out: xs1 [16][32][109][109]
// ===========================================================================
__global__ void loc_conv1_pool(const float* __restrict__ img,
                               const float* __restrict__ w,   // [32][3][7][7]
                               const float* __restrict__ b,
                               float* __restrict__ out) {
    const int W2 = 109, HW2 = 109 * 109;
    int o = blockIdx.x * blockDim.x + threadIdx.x;
    int px = o % W2;
    int t  = o / W2;
    int py = t % W2;  t /= W2;
    int c  = t % 32;
    int n  = t / 32;

    float a00 = b[c], a01 = b[c], a10 = b[c], a11 = b[c];
    int iy0 = 2 * py, ix0 = 2 * px;
    for (int ci = 0; ci < 3; ++ci) {
        const float* im = img + ((size_t)(n * 3 + ci) * 224 + iy0) * 224 + ix0;
        float win[8][8];
        #pragma unroll
        for (int r = 0; r < 8; ++r)
            #pragma unroll
            for (int s = 0; s < 8; ++s)
                win[r][s] = im[r * 224 + s];
        const float* wb = w + (size_t)(c * 3 + ci) * 49;
        #pragma unroll
        for (int ky = 0; ky < 7; ++ky)
            #pragma unroll
            for (int kx = 0; kx < 7; ++kx) {
                float wv = wb[ky * 7 + kx];
                a00 = fmaf(wv, win[ky][kx],         a00);
                a01 = fmaf(wv, win[ky][kx + 1],     a01);
                a10 = fmaf(wv, win[ky + 1][kx],     a10);
                a11 = fmaf(wv, win[ky + 1][kx + 1], a11);
            }
    }
    float m = fmaxf(fmaxf(a00, a01), fmaxf(a10, a11));
    out[(size_t)(n * 32 + c) * HW2 + py * W2 + px] = fmaxf(m, 0.0f);
}

// ===========================================================================
// Localization conv2: 5x5 VALID (32 -> 64) + maxpool2 + relu
// in : xs1 [16][32][109][109]    out: xs2 [16][64][52][52]
// ===========================================================================
__global__ void loc_conv2_pool(const float* __restrict__ x,
                               const float* __restrict__ w,   // [64][32][5][5]
                               const float* __restrict__ b,
                               float* __restrict__ out) {
    const int W1 = 109, W2 = 52, HW2 = 52 * 52;
    int o = blockIdx.x * blockDim.x + threadIdx.x;
    int px = o % W2;
    int t  = o / W2;
    int py = t % W2;  t /= W2;
    int c  = t % 64;
    int n  = t / 64;

    float a00 = b[c], a01 = b[c], a10 = b[c], a11 = b[c];
    int iy0 = 2 * py, ix0 = 2 * px;
    for (int ci = 0; ci < 32; ++ci) {
        const float* im = x + ((size_t)(n * 32 + ci) * W1 + iy0) * W1 + ix0;
        float win[6][6];
        #pragma unroll
        for (int r = 0; r < 6; ++r)
            #pragma unroll
            for (int s = 0; s < 6; ++s)
                win[r][s] = im[r * W1 + s];
        const float* wb = w + (size_t)(c * 32 + ci) * 25;
        #pragma unroll
        for (int ky = 0; ky < 5; ++ky)
            #pragma unroll
            for (int kx = 0; kx < 5; ++kx) {
                float wv = wb[ky * 5 + kx];
                a00 = fmaf(wv, win[ky][kx],         a00);
                a01 = fmaf(wv, win[ky][kx + 1],     a01);
                a10 = fmaf(wv, win[ky + 1][kx],     a10);
                a11 = fmaf(wv, win[ky + 1][kx + 1], a11);
            }
    }
    float m = fmaxf(fmaxf(a00, a01), fmaxf(a10, a11));
    out[(size_t)(n * 64 + c) * HW2 + py * W2 + px] = fmaxf(m, 0.0f);
}

// ===========================================================================
// fc1: h[n][j] = dot(xs2[n,:], fc1_w[j,:]) + fc1_b[j]   (K = 173056)
// one block per (n,j)
// ===========================================================================
__global__ void fc1_kernel(const float* __restrict__ xs,
                           const float* __restrict__ w,
                           const float* __restrict__ b,
                           float* __restrict__ h) {
    const int K = 173056;
    int n = blockIdx.x >> 4;
    int j = blockIdx.x & 15;
    const float* xr = xs + (size_t)n * K;
    const float* wr = w  + (size_t)j * K;
    float p = 0.0f;
    for (int i = threadIdx.x; i < K; i += 256)
        p = fmaf(xr[i], wr[i], p);
    __shared__ float red[256];
    red[threadIdx.x] = p;
    __syncthreads();
    for (int s = 128; s > 0; s >>= 1) {
        if (threadIdx.x < s) red[threadIdx.x] += red[threadIdx.x + s];
        __syncthreads();
    }
    if (threadIdx.x == 0) h[n * 16 + j] = red[0] + b[j];
}

// ===========================================================================
// theta: theta = h @ fc2_w^T + fc2_b ; also 2x3 affine inverse
// ===========================================================================
__global__ void theta_kernel(const float* __restrict__ h,
                             const float* __restrict__ fc2w,  // [6][16]
                             const float* __restrict__ fc2b,
                             float* __restrict__ theta,       // [16][6]
                             float* __restrict__ theta_inv) { // [16][6]
    int n = threadIdx.x;
    if (n >= 16) return;
    float t[6];
    #pragma unroll
    for (int o = 0; o < 6; ++o) {
        float acc = fc2b[o];
        #pragma unroll
        for (int k = 0; k < 16; ++k)
            acc = fmaf(fc2w[o * 16 + k], h[n * 16 + k], acc);
        t[o] = acc;
        theta[n * 6 + o] = acc;
    }
    float a = t[0], bb = t[1], c = t[2], d = t[3], e = t[4], f = t[5];
    float id = 1.0f / (a * e - bb * d);
    theta_inv[n * 6 + 0] =  e * id;
    theta_inv[n * 6 + 1] = -bb * id;
    theta_inv[n * 6 + 2] = (bb * f - c * e) * id;
    theta_inv[n * 6 + 3] = -d * id;
    theta_inv[n * 6 + 4] =  a * id;
    theta_inv[n * 6 + 5] = (c * d - a * f) * id;
}

// ===========================================================================
// Bilinear grid sample helpers
// ===========================================================================
struct Taps { int o00, o01, o10, o11; float w00, w01, w10, w11; };

__device__ __forceinline__ Taps make_taps(const float* th, int h, int w) {
    float gx = (2.0f * w + 1.0f) * (1.0f / 224.0f) - 1.0f;
    float gy = (2.0f * h + 1.0f) * (1.0f / 224.0f) - 1.0f;
    float sx = th[0] * gx + th[1] * gy + th[2];
    float sy = th[3] * gx + th[4] * gy + th[5];
    float ix = (sx + 1.0f) * 112.0f - 0.5f;
    float iy = (sy + 1.0f) * 112.0f - 0.5f;
    float x0f = floorf(ix), y0f = floorf(iy);
    float dx = ix - x0f, dy = iy - y0f;
    int x0 = (int)x0f, y0 = (int)y0f;
    int x1 = x0 + 1, y1 = y0 + 1;
    float vx0 = (x0 >= 0 && x0 < 224) ? 1.0f : 0.0f;
    float vx1 = (x1 >= 0 && x1 < 224) ? 1.0f : 0.0f;
    float vy0 = (y0 >= 0 && y0 < 224) ? 1.0f : 0.0f;
    float vy1 = (y1 >= 0 && y1 < 224) ? 1.0f : 0.0f;
    int x0c = min(max(x0, 0), 223), x1c = min(max(x1, 0), 223);
    int y0c = min(max(y0, 0), 223), y1c = min(max(y1, 0), 223);
    Taps tp;
    tp.w00 = (1.0f - dy) * (1.0f - dx) * vy0 * vx0;
    tp.w01 = (1.0f - dy) * dx          * vy0 * vx1;
    tp.w10 = dy * (1.0f - dx)          * vy1 * vx0;
    tp.w11 = dy * dx                   * vy1 * vx1;
    tp.o00 = y0c * 224 + x0c;  tp.o01 = y0c * 224 + x1c;
    tp.o10 = y1c * 224 + x0c;  tp.o11 = y1c * 224 + x1c;
    return tp;
}

// grid_sample(ori_img, grid(theta)) -> bf16, C=3
__global__ void grid_sample_bf16(const float* __restrict__ img,
                                 const float* __restrict__ theta,
                                 ushort_t* __restrict__ out) {
    const int HW = 224 * 224;
    int p = blockIdx.x * blockDim.x + threadIdx.x;   // over 16*224*224
    int w = p % 224;
    int t = p / 224;
    int h = t % 224;
    int n = t / 224;
    Taps tp = make_taps(theta + n * 6, h, w);
    #pragma unroll
    for (int c = 0; c < 3; ++c) {
        const float* base = img + (size_t)(n * 3 + c) * HW;
        float v = tp.w00 * base[tp.o00] + tp.w01 * base[tp.o01]
                + tp.w10 * base[tp.o10] + tp.w11 * base[tp.o11];
        out[(size_t)(n * 3 + c) * HW + h * 224 + w] = f2bf(v);
    }
}

// grid_sample(Y, grid(theta_inv)) -> f32, C=64  (final output)
__global__ void grid_sample_f32(const float* __restrict__ img,
                                const float* __restrict__ theta,
                                float* __restrict__ out) {
    const int HW = 224 * 224;
    int p = blockIdx.x * blockDim.x + threadIdx.x;
    int w = p % 224;
    int t = p / 224;
    int h = t % 224;
    int n = t / 224;
    Taps tp = make_taps(theta + n * 6, h, w);
    for (int c = 0; c < 64; ++c) {
        const float* base = img + (size_t)(n * 64 + c) * HW;
        float v = tp.w00 * base[tp.o00] + tp.w01 * base[tp.o01]
                + tp.w10 * base[tp.o10] + tp.w11 * base[tp.o11];
        out[(size_t)(n * 64 + c) * HW + h * 224 + w] = v;
    }
}

// ===========================================================================
// Weight convert: OIHW f32 -> padded bf16  Wt[64][Kpad]
// ===========================================================================
__global__ void wconvert(const float* __restrict__ w, ushort_t* __restrict__ wt,
                         int K, int Kpad) {
    int idx = blockIdx.x * blockDim.x + threadIdx.x;
    if (idx >= Kpad * 64) return;
    int k  = idx >> 6;
    int co = idx & 63;
    float v = (k < K) ? w[(size_t)co * K + k] : 0.0f;
    wt[(size_t)co * Kpad + k] = f2bf(v);
}

// ===========================================================================
// Implicit-GEMM 3x3 pad-1 conv via v_wmma_f32_16x16x32_bf16.
//   in  : bf16 [16][Cin][224][224]
//   wt  : bf16 [64][Kpad]  (cout-major weights, K = Cin*9 padded to Kpad)
//   out : f32  [16][64][224][224]  (conv + bias, no activation)
//
// Workgroup = 128 threads = 4 waves; covers 4 adjacent rows x 16 pixels.
// Wave w owns row y0+w and computes ALL 64 couts (4 accumulator tiles),
// reusing its A fragment 4x per k-step:
//     per k-step: 2 ds_load_b128 (A) + 8 global_load_b128 (B) + 4 WMMA
// B reads per lane are 8 16-B pieces of Wt[cout][k], 2 for each of 4 couts;
// across the 32 lanes of a k-step that is a 4-KB, 64-row x 64-B strided, L2-resident
// chunk (full bf16 weight matrix is only 73 KB).
//
// Fragment layouts (ISA 7.12.2):
//   A 16x32 bf16 : lane&15 = row M, lane>>4 selects K-halves {0..7,16..23} /
//                  {8..15,24..31}; vector elt i -> K pair (2i,2i+1).
//   B 32x16 bf16 : lane&15 = column N, lane>>4 and elt select K as for A.
//   C/D 16x16 f32: lane&15 = N, VGPR r -> M = r + 8*(lane>>4).
// ===========================================================================
__global__ void conv3x3_wmma(const ushort_t* __restrict__ in,
                             const ushort_t* __restrict__ wt,
                             const float* __restrict__ bias,
                             float* __restrict__ out,
                             int Cin, int Kpad) {
    extern __shared__ ushort_t Als[];            // [4][16][Kpad]
    const int H = 224, W = 224, HW = H * W;
    int blk = blockIdx.x;                        // 16 * 56 * 14 blocks
    int xs0 = (blk % 14) * 16;
    int t   = blk / 14;
    int y0  = (t % 56) * 4;
    int n   = t / 56;
    int K = Cin * 9;
    const int rowK = 16 * Kpad;                  // LDS elems per strip row

    // ---- build im2col for 4 rows x 16 pixels: Als[row][px][k] ----
    for (int idx = threadIdx.x; idx < 4 * rowK; idx += blockDim.x) {
        int row = idx / rowK;
        int rem = idx - row * rowK;
        int px  = rem / Kpad;
        int k   = rem - px * Kpad;
        ushort_t v = 0;
        if (k < K) {
            int cin = k / 9;
            int r   = k - cin * 9;
            int dy  = r / 3, dx = r - dy * 3;
            int yy = y0 + row + dy - 1;
            int xx = xs0 + px + dx - 1;
            if (yy >= 0 && yy < H && xx >= 0 && xx < W)
                v = in[((size_t)(n * Cin + cin) * H + yy) * W + xx];
        }
        Als[idx] = v;
    }
    __syncthreads();

    int wave = threadIdx.x >> 5;                 // row within block
    int lane = threadIdx.x & 31;
    int arow = lane & 15;                        // A fragment row (pixel)
    int hi   = lane >> 4;

    union Frag { v16bf v; u32x4 q[2]; };
    v8f acc0 = {}, acc1 = {}, acc2 = {}, acc3 = {};

    const ushort_t* abase = Als + wave * rowK + arow * Kpad;
    for (int ks = 0; ks < Kpad; ks += 32) {
        Frag a;
        a.q[0] = *(const u32x4*)(abase + ks + hi * 8);
        a.q[1] = *(const u32x4*)(abase + ks + 16 + hi * 8);
        const ushort_t* br = wt + (size_t)arow * Kpad + ks + hi * 8;   // Wt[cout][k], same K order as A
        const size_t bt = (size_t)16 * Kpad;
        Frag b0, b1, b2, b3;
        b0.q[0] = *(const u32x4*)(br);          b0.q[1] = *(const u32x4*)(br + 16);
        b1.q[0] = *(const u32x4*)(br + bt);     b1.q[1] = *(const u32x4*)(br + bt + 16);
        b2.q[0] = *(const u32x4*)(br + 2 * bt); b2.q[1] = *(const u32x4*)(br + 2 * bt + 16);
        b3.q[0] = *(const u32x4*)(br + 3 * bt); b3.q[1] = *(const u32x4*)(br + 3 * bt + 16);
        acc0 = __builtin_amdgcn_wmma_f32_16x16x32_bf16(
                   false, a.v, false, b0.v, (short)0, acc0, false, false);
        acc1 = __builtin_amdgcn_wmma_f32_16x16x32_bf16(
                   false, a.v, false, b1.v, (short)0, acc1, false, false);
        acc2 = __builtin_amdgcn_wmma_f32_16x16x32_bf16(
                   false, a.v, false, b2.v, (short)0, acc2, false, false);
        acc3 = __builtin_amdgcn_wmma_f32_16x16x32_bf16(
                   false, a.v, false, b3.v, (short)0, acc3, false, false);
    }

    int nn = lane & 15;
    int y  = y0 + wave;
    size_t pixbase = (size_t)y * W + xs0;
    #pragma unroll
    for (int j = 0; j < 4; ++j) {
        int cout = j * 16 + nn;
        float bv = bias[cout];
        float* ob = out + (size_t)(n * 64 + cout) * HW + pixbase;
        v8f acc = (j == 0) ? acc0 : (j == 1) ? acc1 : (j == 2) ? acc2 : acc3;
        #pragma unroll
        for (int r = 0; r < 8; ++r) {
            int m = r + 8 * hi;                  // pixel within strip
            ob[m] = acc[r] + bv;
        }
    }
}

// ===========================================================================
// BN stats: per-channel mean & rstd over N*H*W = 802816 elems, NCHW
// ===========================================================================
__global__ void bn_stats(const float* __restrict__ Y,
                         float* __restrict__ mean,
                         float* __restrict__ rstd) {
    const int HW = 224 * 224, TOT = 16 * HW;
    int c = blockIdx.x;
    float s = 0.0f, q = 0.0f;
    for (int idx = threadIdx.x; idx < TOT; idx += 256) {
        int nn = idx / HW;
        int p  = idx - nn * HW;
        float v = Y[(size_t)(nn * 64 + c) * HW + p];
        s += v;
        q = fmaf(v, v, q);
    }
    __shared__ float rs[256], rq[256];
    rs[threadIdx.x] = s; rq[threadIdx.x] = q;
    __syncthreads();
    for (int st = 128; st > 0; st >>= 1) {
        if (threadIdx.x < st) {
            rs[threadIdx.x] += rs[threadIdx.x + st];
            rq[threadIdx.x] += rq[threadIdx.x + st];
        }
        __syncthreads();
    }
    if (threadIdx.x == 0) {
        float m = rs[0] * (1.0f / TOT);
        float var = rq[0] * (1.0f / TOT) - m * m;
        mean[c] = m;
        rstd[c] = rsqrtf(var + 1e-5f);
    }
}

// normalize + relu -> bf16 copy (feeds next WMMA conv)
__global__ void bn_apply_bf16(const float* __restrict__ Y,
                              const float* __restrict__ mean,
                              const float* __restrict__ rstd,
                              const float* __restrict__ g,
                              const float* __restrict__ b,
                              ushort_t* __restrict__ out) {
    const int HW = 224 * 224;
    int i = blockIdx.x * blockDim.x + threadIdx.x;
    int c = (i / HW) & 63;
    float v = (Y[i] - mean[c]) * rstd[c] * g[c] + b[c];
    out[i] = f2bf(fmaxf(v, 0.0f));
}

// normalize + relu in place (f32)
__global__ void bn_apply_f32(float* __restrict__ Y,
                             const float* __restrict__ mean,
                             const float* __restrict__ rstd,
                             const float* __restrict__ g,
                             const float* __restrict__ b) {
    const int HW = 224 * 224;
    int i = blockIdx.x * blockDim.x + threadIdx.x;
    int c = (i / HW) & 63;
    float v = (Y[i] - mean[c]) * rstd[c] * g[c] + b[c];
    Y[i] = fmaxf(v, 0.0f);
}

// ===========================================================================
// Host driver
// ===========================================================================
static size_t align256(size_t x) { return (x + 255) & ~(size_t)255; }

extern "C" void kernel_launch(void* const* d_in, const int* in_sizes, int n_in,
                              void* d_out, int out_size, void* d_ws, size_t ws_size,
                              hipStream_t stream) {
    const float* ori    = (const float*)d_in[0];
    const float* lw1    = (const float*)d_in[1];
    const float* lb1    = (const float*)d_in[2];
    const float* lw2    = (const float*)d_in[3];
    const float* lb2    = (const float*)d_in[4];
    const float* fc1w   = (const float*)d_in[5];
    const float* fc1b   = (const float*)d_in[6];
    const float* fc2w   = (const float*)d_in[7];
    const float* fc2b   = (const float*)d_in[8];
    const float* cw1    = (const float*)d_in[9];
    const float* cb1    = (const float*)d_in[10];
    const float* bg1    = (const float*)d_in[11];
    const float* bb1    = (const float*)d_in[12];
    const float* cw2    = (const float*)d_in[13];
    const float* cb2    = (const float*)d_in[14];
    const float* bg2    = (const float*)d_in[15];
    const float* bb2    = (const float*)d_in[16];
    float* outp = (float*)d_out;

    const size_t XS1 = (size_t)16 * 32 * 109 * 109;  // 6,083,072
    const size_t XS2 = (size_t)16 * 64 * 52 * 52;    // 2,768,896
    const size_t SMP = (size_t)16 * 3 * 224 * 224;   // 2,408,448
    const size_t YE  = (size_t)16 * 64 * 224 * 224;  // 51,380,224

    char* ws = (char*)d_ws;
    size_t off = 0;
    float*    xs1    = (float*)(ws + off);  off = align256(off + XS1 * 4);
    float*    xs2    = (float*)(ws + off);  off = align256(off + XS2 * 4);
    float*    hbuf   = (float*)(ws + off);  off = align256(off + 256 * 4);
    float*    theta  = (float*)(ws + off);  off = align256(off + 96 * 4);
    float*    thinv  = (float*)(ws + off);  off = align256(off + 96 * 4);
    float*    stats  = (float*)(ws + off);  off = align256(off + 256 * 4);  // m1,r1,m2,r2
    ushort_t* w1t    = (ushort_t*)(ws + off); off = align256(off + (size_t)32 * 64 * 2);
    ushort_t* w2t    = (ushort_t*)(ws + off); off = align256(off + (size_t)576 * 64 * 2);
    ushort_t* sampbf = (ushort_t*)(ws + off); off = align256(off + SMP * 2);
    ushort_t* x1bf   = (ushort_t*)(ws + off); off = align256(off + YE * 2);
    float*    Y      = (float*)(ws + off);    off = align256(off + YE * 4);
    (void)ws_size; (void)n_in; (void)in_sizes; (void)out_size;

    // 1) localization network
    loc_conv1_pool<<<(int)(XS1 / 256), 256, 0, stream>>>(ori, lw1, lb1, xs1);
    loc_conv2_pool<<<(int)(XS2 / 256), 256, 0, stream>>>(xs1, lw2, lb2, xs2);
    fc1_kernel<<<256, 256, 0, stream>>>(xs2, fc1w, fc1b, hbuf);
    theta_kernel<<<1, 32, 0, stream>>>(hbuf, fc2w, fc2b, theta, thinv);

    // 2) forward grid sample (fp32 img -> bf16)
    grid_sample_bf16<<<3136, 256, 0, stream>>>(ori, theta, sampbf);

    // 3) weight pad+convert for WMMA convs
    wconvert<<<8, 256, 0, stream>>>(cw1, w1t, 27, 32);
    wconvert<<<144, 256, 0, stream>>>(cw2, w2t, 576, 576);

    // 4 rows x 16 px per workgroup: 16 * (224/4) * (224/16) = 12544 blocks
    const int blocks = 16 * 56 * 14;

    // 4) main conv1 (Cin=3, Kpad=32) -> Y, then BN stats + apply -> bf16 x1
    conv3x3_wmma<<<blocks, 128, 4 * 16 * 32 * 2, stream>>>(sampbf, w1t, cb1, Y, 3, 32);
    bn_stats<<<64, 256, 0, stream>>>(Y, stats, stats + 64);
    bn_apply_bf16<<<(int)(YE / 256), 256, 0, stream>>>(Y, stats, stats + 64,
                                                       bg1, bb1, x1bf);

    // 5) main conv2 (Cin=64, Kpad=576) -> Y (reused), BN stats + apply in place
    conv3x3_wmma<<<blocks, 128, 4 * 16 * 576 * 2, stream>>>(x1bf, w2t, cb2, Y, 64, 576);
    bn_stats<<<64, 256, 0, stream>>>(Y, stats + 128, stats + 192);
    bn_apply_f32<<<(int)(YE / 256), 256, 0, stream>>>(Y, stats + 128, stats + 192,
                                                      bg2, bb2);

    // 6) inverse grid sample -> final output
    grid_sample_f32<<<3136, 256, 0, stream>>>(Y, thinv, outp);
}
